// SAGE_68839735820559
// MI455X (gfx1250) — matrix operand order repaired and checked by reading, not compiled, not run on hardware
//
#include <hip/hip_runtime.h>
#include <stdint.h>

// ---------------------------------------------------------------------------
// SAGE-style 2-layer GNN forward for MI455X (gfx1250, wave32, WMMA).
//   t = prelu( t@W1^T + b1 + spmm(t)@W2^T + b2 )   (x2 layers)
// Dense GEMMs via v_wmma_f32_16x16x32_bf16 (bf16 operands, f32 accum).
// SpMM via edge-parallel gather + global_atomic_add_f32 (L2-resident table).
// ---------------------------------------------------------------------------

typedef __attribute__((ext_vector_type(16))) __bf16 v16bf;
typedef __attribute__((ext_vector_type(8)))  float  v8f;

#define N_NODES 100000
#define N_EDGES 1600000
#define DIM     128

__device__ __forceinline__ unsigned int f2bf_bits(float f) {
  unsigned int u = __float_as_uint(f);
  // round-to-nearest-even f32 -> bf16
  return (u + 0x7FFFu + ((u >> 16) & 1u)) >> 16;
}
__device__ __forceinline__ unsigned int pack2bf(float a, float b) {
  return f2bf_bits(a) | (f2bf_bits(b) << 16);
}
__device__ __forceinline__ float bf2f(unsigned short h) {
  return __uint_as_float(((unsigned int)h) << 16);
}

// ---------------------------------------------------------------------------
// Zero-fill f32 accumulator (float4 stores).
// ---------------------------------------------------------------------------
__global__ void sage_zero_kernel(float4* __restrict__ p, int n4) {
  int i = blockIdx.x * blockDim.x + threadIdx.x;
  if (i < n4) p[i] = make_float4(0.f, 0.f, 0.f, 0.f);
}

// ---------------------------------------------------------------------------
// COO SpMM: S[row[e], :] += val[e] * src[col[e], :]
// One wave32 per edge; each lane handles 4 consecutive features.
// src is f32 (layer 0: X) or bf16 (layer 1: T1). Accumulation is f32 atomics.
// ---------------------------------------------------------------------------
template <bool SRC_BF16>
__global__ void sage_spmm_kernel(const void* __restrict__ src,
                                 const int* __restrict__ row,
                                 const int* __restrict__ col,
                                 const float* __restrict__ val,
                                 float* __restrict__ S, int nE) {
  const int e = blockIdx.x * 8 + (threadIdx.x >> 5);
  if (e >= nE) return;
  const int lane = threadIdx.x & 31;

  const int   r = row[e];
  const int   c = col[e];
  const float v = val[e];

  float f0, f1, f2, f3;
  if (SRC_BF16) {
    const unsigned short* p = (const unsigned short*)src + (size_t)c * DIM + lane * 4;
    ushort4 u = *(const ushort4*)p;                    // 8B gather (bf16 halves L2 traffic)
    f0 = bf2f(u.x); f1 = bf2f(u.y); f2 = bf2f(u.z); f3 = bf2f(u.w);
  } else {
    const float* p = (const float*)src + (size_t)c * DIM + lane * 4;
    float4 f = *(const float4*)p;                      // 16B gather, L2-resident table
    f0 = f.x; f1 = f.y; f2 = f.z; f3 = f.w;
  }

  float* out = S + (size_t)r * DIM + lane * 4;
  __hip_atomic_fetch_add(out + 0, v * f0, __ATOMIC_RELAXED, __HIP_MEMORY_SCOPE_AGENT);
  __hip_atomic_fetch_add(out + 1, v * f1, __ATOMIC_RELAXED, __HIP_MEMORY_SCOPE_AGENT);
  __hip_atomic_fetch_add(out + 2, v * f2, __ATOMIC_RELAXED, __HIP_MEMORY_SCOPE_AGENT);
  __hip_atomic_fetch_add(out + 3, v * f3, __ATOMIC_RELAXED, __HIP_MEMORY_SCOPE_AGENT);
}

// ---------------------------------------------------------------------------
// Fused layer GEMM:
//   out[n,h] = prelu( sum_k A[n,k]*W1[h,k] + sum_k S[n,k]*W2[h,k] + b1[h]+b2[h] )
// i.e. a K=256 GEMM with A' = [A | S] and B' = [W1 ; W2] stacked along K.
// Block = 128 threads (4 wave32s), output tile 64 rows x 128 cols.
// Wave w -> columns [32w, 32w+32): 4 M-tiles x 2 N-tiles of 16x16 WMMA.
// Operands staged to LDS as bf16, pitch 40 (16B aligned b128 fragment reads).
// ---------------------------------------------------------------------------
__launch_bounds__(128)
__global__ void sage_fused_layer_kernel(const float* __restrict__ Af,
                                        const unsigned short* __restrict__ Ab,
                                        int a_is_bf,
                                        const float* __restrict__ S,
                                        const float* __restrict__ W1,
                                        const float* __restrict__ W2,
                                        const float* __restrict__ b1,
                                        const float* __restrict__ b2,
                                        const float* __restrict__ a_ptr,
                                        float* __restrict__ outF,
                                        unsigned short* __restrict__ outB,
                                        int write_f, int write_b) {
  __shared__ alignas(16) unsigned short ldsA[64 * 40];    // 64 rows  x 32 K (pitch 40)
  __shared__ alignas(16) unsigned short ldsB[128 * 40];   // 128 cols x 32 K (pitch 40)

  const int tid      = threadIdx.x;
  const int lane     = tid & 31;
  const int wave     = tid >> 5;
  const int lmod     = lane & 15;
  const int lhalf    = lane >> 4;       // 0 or 1
  const int rowBlock = blockIdx.x * 64;

  v8f acc[4][2];
#pragma unroll
  for (int mt = 0; mt < 4; ++mt)
#pragma unroll
    for (int nt = 0; nt < 2; ++nt) acc[mt][nt] = {};

  for (int s = 0; s < 8; ++s) {         // K = 256 in steps of 32
    const int k0 = s * 32;

    // ---- stage A' tile (64x32 bf16): k<128 -> A, k>=128 -> S ----
    {
      const int r    = tid >> 1;
      const int kb   = (tid & 1) * 16;
      const int grow = rowBlock + r;
      unsigned int pk[8];
      if (grow < N_NODES) {
        if (k0 >= 128) {
          const float4* p = (const float4*)(S + (size_t)grow * DIM + (k0 - 128) + kb);
#pragma unroll
          for (int q = 0; q < 4; ++q) {
            float4 f = p[q];
            pk[2 * q]     = pack2bf(f.x, f.y);
            pk[2 * q + 1] = pack2bf(f.z, f.w);
          }
        } else if (a_is_bf) {
          const uint4* p = (const uint4*)(Ab + (size_t)grow * DIM + k0 + kb);
          uint4 q0 = p[0], q1 = p[1];
          pk[0] = q0.x; pk[1] = q0.y; pk[2] = q0.z; pk[3] = q0.w;
          pk[4] = q1.x; pk[5] = q1.y; pk[6] = q1.z; pk[7] = q1.w;
        } else {
          const float4* p = (const float4*)(Af + (size_t)grow * DIM + k0 + kb);
#pragma unroll
          for (int q = 0; q < 4; ++q) {
            float4 f = p[q];
            pk[2 * q]     = pack2bf(f.x, f.y);
            pk[2 * q + 1] = pack2bf(f.z, f.w);
          }
        }
      } else {
#pragma unroll
        for (int q = 0; q < 8; ++q) pk[q] = 0u;
      }
      *(uint4*)&ldsA[r * 40 + kb]     = make_uint4(pk[0], pk[1], pk[2], pk[3]);
      *(uint4*)&ldsA[r * 40 + kb + 8] = make_uint4(pk[4], pk[5], pk[6], pk[7]);
    }

    // ---- stage B' tile (128 cols x 32 K bf16), column-major in LDS ----
    // B'[k,h] = Wsel[h,k]  => one W row per thread, contiguous global read.
    {
      const float* wsrc = (k0 < 128) ? W1 : W2;
      const int    kr   = (k0 < 128) ? k0 : (k0 - 128);
      const float4* p   = (const float4*)(wsrc + (size_t)tid * DIM + kr);
      unsigned int pk[16];
#pragma unroll
      for (int q = 0; q < 8; ++q) {
        float4 f = p[q];
        pk[2 * q]     = pack2bf(f.x, f.y);
        pk[2 * q + 1] = pack2bf(f.z, f.w);
      }
#pragma unroll
      for (int q = 0; q < 4; ++q)
        *(uint4*)&ldsB[tid * 40 + 8 * q] =
            make_uint4(pk[4 * q], pk[4 * q + 1], pk[4 * q + 2], pk[4 * q + 3]);
    }
    __syncthreads();

    // ---- fragments (ISA 7.12.2 16-bit layouts) + 8 WMMA per wave ----
    union Frag { uint4 q[2]; v16bf v; };
    Frag afr[4], bfr[2];
#pragma unroll
    for (int mt = 0; mt < 4; ++mt) {
      // lane<16: K 0-7 / 16-23 ; lane>=16: K 8-15 / 24-31
      const unsigned short* base = &ldsA[(mt * 16 + lmod) * 40 + lhalf * 8];
      afr[mt].q[0] = *(const uint4*)base;
      afr[mt].q[1] = *(const uint4*)(base + 16);
    }
#pragma unroll
    for (int nt = 0; nt < 2; ++nt) {
      // lane<16: K 0-7 / 16-23 ; lane>=16: K 8-15 / 24-31, column = lane&15
      const unsigned short* base = &ldsB[(wave * 32 + nt * 16 + lmod) * 40 + lhalf * 8];
      bfr[nt].q[0] = *(const uint4*)base;
      bfr[nt].q[1] = *(const uint4*)(base + 16);
    }
#pragma unroll
    for (int mt = 0; mt < 4; ++mt)
#pragma unroll
      for (int nt = 0; nt < 2; ++nt)
        acc[mt][nt] = __builtin_amdgcn_wmma_f32_16x16x32_bf16(
            false, afr[mt].v, false, bfr[nt].v, (short)0, acc[mt][nt], false, false);
    __syncthreads();
  }

  // ---- epilogue: bias + PReLU, write f32 and/or bf16 ----
  const float slope = a_ptr[0];
#pragma unroll
  for (int nt = 0; nt < 2; ++nt) {
    const int   c    = wave * 32 + nt * 16 + lmod;
    const float bias = b1[c] + b2[c];
#pragma unroll
    for (int mt = 0; mt < 4; ++mt) {
#pragma unroll
      for (int r = 0; r < 8; ++r) {
        const int grow = rowBlock + mt * 16 + lhalf * 8 + r;   // C/D layout: VGPR r -> M = r + 8*lhalf
        if (grow < N_NODES) {
          float x = acc[mt][nt][r] + bias;
          x = (x >= 0.0f) ? x : slope * x;
          if (write_f) outF[(size_t)grow * DIM + c] = x;
          if (write_b) outB[(size_t)grow * DIM + c] = (unsigned short)f2bf_bits(x);
        }
      }
    }
  }
}

// ---------------------------------------------------------------------------
// Host-side launch sequence (graph-capture safe: only kernel launches).
// Workspace: S (N*128 f32, 51.2MB) | T1 (N*128 bf16, 25.6MB)  => 76.8MB.
// ---------------------------------------------------------------------------
extern "C" void kernel_launch(void* const* d_in, const int* in_sizes, int n_in,
                              void* d_out, int out_size, void* d_ws, size_t ws_size,
                              hipStream_t stream) {
  (void)in_sizes; (void)n_in; (void)out_size; (void)ws_size;

  const float* X    = (const float*)d_in[0];
  const int*   row  = (const int*)d_in[1];
  const int*   col  = (const int*)d_in[2];
  const float* val  = (const float*)d_in[3];
  const float* W1_0 = (const float*)d_in[4];
  const float* b1_0 = (const float*)d_in[5];
  const float* W2_0 = (const float*)d_in[6];
  const float* b2_0 = (const float*)d_in[7];
  const float* a_0  = (const float*)d_in[8];
  const float* W1_1 = (const float*)d_in[9];
  const float* b1_1 = (const float*)d_in[10];
  const float* W2_1 = (const float*)d_in[11];
  const float* b2_1 = (const float*)d_in[12];
  const float* a_1  = (const float*)d_in[13];

  float*          S  = (float*)d_ws;
  unsigned short* T1 = (unsigned short*)((char*)d_ws + (size_t)N_NODES * DIM * sizeof(float));

  const int n4   = N_NODES * DIM / 4;
  const int zblk = (n4 + 255) / 256;
  const int eblk = (N_EDGES + 7) / 8;        // 8 edges (waves) per 256-thread block
  const int gblk = (N_NODES + 63) / 64;      // 64 output rows per block

  // ---- layer 0 ----
  sage_zero_kernel<<<zblk, 256, 0, stream>>>((float4*)S, n4);
  sage_spmm_kernel<false><<<eblk, 256, 0, stream>>>(X, row, col, val, S, N_EDGES);
  sage_fused_layer_kernel<<<gblk, 128, 0, stream>>>(X, nullptr, 0, S,
                                                    W1_0, W2_0, b1_0, b2_0, a_0,
                                                    nullptr, T1, 0, 1);
  // ---- layer 1 ----
  sage_zero_kernel<<<zblk, 256, 0, stream>>>((float4*)S, n4);
  sage_spmm_kernel<true><<<eblk, 256, 0, stream>>>(T1, row, col, val, S, N_EDGES);
  sage_fused_layer_kernel<<<gblk, 128, 0, stream>>>(nullptr, T1, 1, S,
                                                    W1_1, W2_1, b1_1, b2_1, a_1,
                                                    (float*)d_out, nullptr, 1, 0);
}
